// MultilayerGRU_61022895341988
// MI455X (gfx1250) — matrix operand order repaired and checked by reading, not compiled, not run on hardware
//
#include <hip/hip_runtime.h>
#include <hip/hip_bf16.h>

// GRU dims
#define BB 32
#define SS 512
#define II 256
#define HH 1024
#define OO 256

#define NWG 26              // persistent workgroups; 26*8 = 208 waves = max tiles in any phase
#define NWAVES (NWG * 8)

typedef __attribute__((ext_vector_type(16))) __bf16 v16bf;
typedef __attribute__((ext_vector_type(8)))  float  v8f;
typedef __hip_bfloat16 bf16;

// ---- WMMA operand loaders (bf16 16x16x32 layouts per CDNA5 ISA 7.12.2) ----
// A (16x32, 16-bit): lane<16 holds row=lane, K {k0..k0+7} then {k0+16..k0+23};
// lane>=16 holds row=lane-16, K {k0+8..15} then {k0+24..31}. p = row base + k0 + 8*half.
__device__ __forceinline__ v16bf ld_a(const bf16* p) {
  union { uint4 q[2]; v16bf v; } u;
  u.q[0] = *(const uint4*)(p);
  u.q[1] = *(const uint4*)(p + 16);
  return u.v;
}
// B (32x16): lane n (n<16) holds column n, K {k0..k0+7} then {k0+16..k0+23}; lanes 16-31 hold K {k0+8..15} then {k0+24..31}.
// Weights stored transposed [N][K] so this is contiguous. p = col base + k0 + 8*half.
__device__ __forceinline__ v16bf ld_b(const bf16* p) {
  union { uint4 q[2]; v16bf v; } u;
  u.q[0] = *(const uint4*)(p);
  u.q[1] = *(const uint4*)(p + 16);
  return u.v;
}

__device__ __forceinline__ float sigm(float x) { return 1.0f / (1.0f + __expf(-x)); }

// One wave computes a 32x16 output tile (two 16x16 WMMA accumulators) of
// C = [A0|A1] * B, A segments row-major bf16 (32 rows), B transposed bf16 [N][K].
__device__ __forceinline__ void tile_gemm(
    const bf16* A0, long sA0, int K0,
    const bf16* A1, long sA1, int K1,
    const bf16* BT, int ldb, int n0,
    v8f& c0, v8f& c1)
{
  const int lane = threadIdx.x & 31;
  const int hlf  = lane >> 4;
  const int lr   = lane & 15;
  v8f z;
#pragma unroll
  for (int i = 0; i < 8; ++i) z[i] = 0.0f;
  c0 = z; c1 = z;
  const bf16* bB = BT + (long)(n0 + lr) * ldb + 8 * hlf;
  const int Kt = K0 + K1;
#pragma unroll 2
  for (int k0 = 0; k0 < Kt; k0 += 32) {
    const bf16* ab; long s; int ks;
    if (k0 < K0) { ab = A0; s = sA0; ks = k0; }
    else         { ab = A1; s = sA1; ks = k0 - K0; }
    const bf16* ar = ab + (long)lr * s + ks + 8 * hlf;
    v16bf av0 = ld_a(ar);            // rows 0..15
    v16bf av1 = ld_a(ar + 16 * s);   // rows 16..31
    v16bf bv  = ld_b(bB + k0);
    c0 = __builtin_amdgcn_wmma_f32_16x16x32_bf16(false, av0, false, bv, (short)0, c0, false, false);
    c1 = __builtin_amdgcn_wmma_f32_16x16x32_bf16(false, av1, false, bv, (short)0, c1, false, false);
  }
}

// Sense-reversing device-wide barrier (26 WGs are trivially co-resident on MI455X).
__device__ __forceinline__ void gbar(unsigned* cnt, unsigned* gen) {
  __syncthreads();
  if (threadIdx.x == 0) {
    __threadfence();
    unsigned g = __hip_atomic_load(gen, __ATOMIC_ACQUIRE, __HIP_MEMORY_SCOPE_AGENT);
    unsigned a = __hip_atomic_fetch_add(cnt, 1u, __ATOMIC_ACQ_REL, __HIP_MEMORY_SCOPE_AGENT);
    if (a == (unsigned)(NWG - 1)) {
      __hip_atomic_store(cnt, 0u, __ATOMIC_RELAXED, __HIP_MEMORY_SCOPE_AGENT);
      __hip_atomic_fetch_add(gen, 1u, __ATOMIC_ACQ_REL, __HIP_MEMORY_SCOPE_AGENT);
    } else {
      while (__hip_atomic_load(gen, __ATOMIC_ACQUIRE, __HIP_MEMORY_SCOPE_AGENT) == g)
        __builtin_amdgcn_s_sleep(1);
    }
  }
  __syncthreads();
}

// ---- one-time prep kernels ----
// dst[(n)*dstK + kOff + k] = bf16(src[k*ld + colOff + n]); K,N multiples of 32.
__global__ void pack_wT(const float* __restrict__ src, int ld, int colOff,
                        bf16* __restrict__ dst, int dstK, int kOff)
{
  __shared__ float tile[32][33];
  const int k0 = blockIdx.x * 32;
  const int n0 = blockIdx.y * 32;
  const int tx = threadIdx.x, ty = threadIdx.y;
#pragma unroll
  for (int i = 0; i < 32; i += 8)
    tile[ty + i][tx] = src[(long)(k0 + ty + i) * ld + colOff + n0 + tx];
  __syncthreads();
#pragma unroll
  for (int i = 0; i < 32; i += 8)
    dst[(long)(n0 + ty + i) * dstK + kOff + k0 + tx] = __float2bfloat16(tile[tx][ty + i]);
}

__global__ void pack_bf16(const float* __restrict__ src, bf16* __restrict__ dst, int n) {
  int i = blockIdx.x * blockDim.x + threadIdx.x;
  if (i < n) dst[i] = __float2bfloat16(src[i]);
}

__global__ void init_h(const float* __restrict__ h0in,
                       float* h0f, float* h1f, bf16* h0b, bf16* h1b) {
  int i = blockIdx.x * blockDim.x + threadIdx.x;   // [0, BB*HH)
  if (i >= BB * HH) return;
  int b = i >> 10, j = i & (HH - 1);
  float v0 = h0in[(long)b * 2 * HH + j];
  float v1 = h0in[(long)b * 2 * HH + HH + j];
  h0f[i] = v0; h1f[i] = v1;
  h0b[i] = __float2bfloat16(v0);
  h1b[i] = __float2bfloat16(v1);
}

// ---- persistent GRU kernel: 512 steps, 4 global barriers per step ----
__global__ void __launch_bounds__(256)
gru_persistent(const bf16* __restrict__ xb,
               const bf16* __restrict__ WzrT0, const bf16* __restrict__ Wg0xT,
               const bf16* __restrict__ Wh0gT, const bf16* __restrict__ W1zrT,
               const bf16* __restrict__ Wx1gT, const bf16* __restrict__ Wh1gT,
               const bf16* __restrict__ WyT,
               const float* __restrict__ bh0, const float* __restrict__ bh1,
               const float* __restrict__ by,
               float* h0f, float* h1f, bf16* h0b, bf16* h1b,
               float* z0buf, float* xg0, bf16* rh0b,
               float* z1buf, float* xg1, bf16* rh1b,
               float* out, unsigned* bar)
{
  const int gw   = blockIdx.x * 8 + (threadIdx.x >> 5);
  const int lane = threadIdx.x & 31;
  const int hlf  = lane >> 4;
  const int lr   = lane & 15;
  unsigned* cnt = bar;
  unsigned* gen = bar + 1;

  for (int t = 0; t < SS; ++t) {
    // ---------- Phase A: layer0 z/r (K=1280), layer0 xg (K=256), y_{t-1} (K=1024) ----------
    for (int tt = gw; tt < 208; tt += NWAVES) {
      v8f c0, c1;
      if (tt < 128) {
        const int n0 = tt * 16;
        tile_gemm(xb + (long)t * II, (long)SS * II, II, h0b, HH, HH, WzrT0, II + HH, n0, c0, c1);
        const int col = n0 + lr;
        const float bb = bh0[col];
        if (n0 < HH) {                       // z gate
#pragma unroll
          for (int i = 0; i < 8; ++i) {
            const int r = i + 8 * hlf;
            z0buf[r * HH + col]        = sigm(c0[i] + bb);
            z0buf[(r + 16) * HH + col] = sigm(c1[i] + bb);
          }
        } else {                             // r gate -> r*h (bf16 A-operand for B0)
          const int hc = col - HH;
#pragma unroll
          for (int i = 0; i < 8; ++i) {
            const int r = i + 8 * hlf;
            rh0b[r * HH + hc]        = __float2bfloat16(sigm(c0[i] + bb) * h0f[r * HH + hc]);
            rh0b[(r + 16) * HH + hc] = __float2bfloat16(sigm(c1[i] + bb) * h0f[(r + 16) * HH + hc]);
          }
        }
      } else if (tt < 192) {                 // xg0 = x_t @ Wx0g + bg
        const int n0 = (tt - 128) * 16;
        tile_gemm(xb + (long)t * II, (long)SS * II, II, nullptr, 0, 0, Wg0xT, II, n0, c0, c1);
        const int col = n0 + lr;
        const float bb = bh0[2 * HH + col];
#pragma unroll
        for (int i = 0; i < 8; ++i) {
          const int r = i + 8 * hlf;
          xg0[r * HH + col]        = c0[i] + bb;
          xg0[(r + 16) * HH + col] = c1[i] + bb;
        }
      } else if (t > 0) {                    // y_{t-1} = h1 @ Wy + by (overlapped)
        const int n0 = (tt - 192) * 16;
        tile_gemm(h1b, HH, HH, nullptr, 0, 0, WyT, HH, n0, c0, c1);
        const int col = n0 + lr;
        const float bb = by[col];
#pragma unroll
        for (int i = 0; i < 8; ++i) {
          const int r = i + 8 * hlf;
          out[((long)r * SS + (t - 1)) * OO + col]        = c0[i] + bb;
          out[((long)(r + 16) * SS + (t - 1)) * OO + col] = c1[i] + bb;
        }
      }
    }
    gbar(cnt, gen);

    // ---------- Phase B0: g0 = tanh(xg0 + (r*h0)@Wh0g); h0' ----------
    for (int tt = gw; tt < 64; tt += NWAVES) {
      v8f c0, c1;
      const int n0 = tt * 16;
      tile_gemm(rh0b, HH, HH, nullptr, 0, 0, Wh0gT, HH, n0, c0, c1);
      const int col = n0 + lr;
#pragma unroll
      for (int i = 0; i < 8; ++i) {
        int r = i + 8 * hlf;
        {
          const float g  = tanhf(xg0[r * HH + col] + c0[i]);
          const float zz = z0buf[r * HH + col];
          const float hn = zz * h0f[r * HH + col] + (1.0f - zz) * g;
          h0f[r * HH + col] = hn;
          h0b[r * HH + col] = __float2bfloat16(hn);
        }
        r += 16;
        {
          const float g  = tanhf(xg0[r * HH + col] + c1[i]);
          const float zz = z0buf[r * HH + col];
          const float hn = zz * h0f[r * HH + col] + (1.0f - zz) * g;
          h0f[r * HH + col] = hn;
          h0b[r * HH + col] = __float2bfloat16(hn);
        }
      }
    }
    gbar(cnt, gen);

    // ---------- Phase A1: layer1 z/r (K=2048), layer1 xg (K=1024) ----------
    for (int tt = gw; tt < 192; tt += NWAVES) {
      v8f c0, c1;
      if (tt < 128) {
        const int n0 = tt * 16;
        tile_gemm(h0b, HH, HH, h1b, HH, HH, W1zrT, 2 * HH, n0, c0, c1);
        const int col = n0 + lr;
        const float bb = bh1[col];
        if (n0 < HH) {
#pragma unroll
          for (int i = 0; i < 8; ++i) {
            const int r = i + 8 * hlf;
            z1buf[r * HH + col]        = sigm(c0[i] + bb);
            z1buf[(r + 16) * HH + col] = sigm(c1[i] + bb);
          }
        } else {
          const int hc = col - HH;
#pragma unroll
          for (int i = 0; i < 8; ++i) {
            const int r = i + 8 * hlf;
            rh1b[r * HH + hc]        = __float2bfloat16(sigm(c0[i] + bb) * h1f[r * HH + hc]);
            rh1b[(r + 16) * HH + hc] = __float2bfloat16(sigm(c1[i] + bb) * h1f[(r + 16) * HH + hc]);
          }
        }
      } else {
        const int n0 = (tt - 128) * 16;
        tile_gemm(h0b, HH, HH, nullptr, 0, 0, Wx1gT, HH, n0, c0, c1);
        const int col = n0 + lr;
        const float bb = bh1[2 * HH + col];
#pragma unroll
        for (int i = 0; i < 8; ++i) {
          const int r = i + 8 * hlf;
          xg1[r * HH + col]        = c0[i] + bb;
          xg1[(r + 16) * HH + col] = c1[i] + bb;
        }
      }
    }
    gbar(cnt, gen);

    // ---------- Phase B1: g1; h1' ----------
    for (int tt = gw; tt < 64; tt += NWAVES) {
      v8f c0, c1;
      const int n0 = tt * 16;
      tile_gemm(rh1b, HH, HH, nullptr, 0, 0, Wh1gT, HH, n0, c0, c1);
      const int col = n0 + lr;
#pragma unroll
      for (int i = 0; i < 8; ++i) {
        int r = i + 8 * hlf;
        {
          const float g  = tanhf(xg1[r * HH + col] + c0[i]);
          const float zz = z1buf[r * HH + col];
          const float hn = zz * h1f[r * HH + col] + (1.0f - zz) * g;
          h1f[r * HH + col] = hn;
          h1b[r * HH + col] = __float2bfloat16(hn);
        }
        r += 16;
        {
          const float g  = tanhf(xg1[r * HH + col] + c1[i]);
          const float zz = z1buf[r * HH + col];
          const float hn = zz * h1f[r * HH + col] + (1.0f - zz) * g;
          h1f[r * HH + col] = hn;
          h1b[r * HH + col] = __float2bfloat16(hn);
        }
      }
    }
    gbar(cnt, gen);
  }

  // final readout y_{S-1}
  for (int tt = gw; tt < 16; tt += NWAVES) {
    v8f c0, c1;
    const int n0 = tt * 16;
    tile_gemm(h1b, HH, HH, nullptr, 0, 0, WyT, HH, n0, c0, c1);
    const int col = n0 + lr;
    const float bb = by[col];
#pragma unroll
    for (int i = 0; i < 8; ++i) {
      const int r = i + 8 * hlf;
      out[((long)r * SS + (SS - 1)) * OO + col]        = c0[i] + bb;
      out[((long)(r + 16) * SS + (SS - 1)) * OO + col] = c1[i] + bb;
    }
  }
  // hidden_state output [B, L=2, H]
  float* hs = out + (long)BB * SS * OO;
  for (int i = blockIdx.x * 256 + threadIdx.x; i < BB * HH; i += NWG * 256) {
    const int b = i >> 10, j = i & (HH - 1);
    hs[(long)b * 2 * HH + j]      = h0f[i];
    hs[(long)b * 2 * HH + HH + j] = h1f[i];
  }
}

extern "C" void kernel_launch(void* const* d_in, const int* in_sizes, int n_in,
                              void* d_out, int out_size, void* d_ws, size_t ws_size,
                              hipStream_t stream) {
  (void)in_sizes; (void)n_in; (void)out_size; (void)ws_size;
  const float* x   = (const float*)d_in[0];
  const float* h0  = (const float*)d_in[1];
  const float* Wx0 = (const float*)d_in[2];
  const float* Wh0 = (const float*)d_in[3];
  const float* bh0 = (const float*)d_in[4];
  const float* Wx1 = (const float*)d_in[5];
  const float* Wh1 = (const float*)d_in[6];
  const float* bh1 = (const float*)d_in[7];
  const float* Wy  = (const float*)d_in[8];
  const float* by  = (const float*)d_in[9];
  float* out = (float*)d_out;

  char* p = (char*)d_ws;
  auto alloc = [&](size_t bytes) { char* r = p; p += (bytes + 255) & ~(size_t)255; return r; };
  unsigned* bar  = (unsigned*)alloc(256);
  bf16* WzrT0 = (bf16*)alloc((size_t)2048 * 1280 * 2);
  bf16* Wg0xT = (bf16*)alloc((size_t)1024 * 256 * 2);
  bf16* Wh0gT = (bf16*)alloc((size_t)1024 * 1024 * 2);
  bf16* W1zrT = (bf16*)alloc((size_t)2048 * 2048 * 2);
  bf16* Wx1gT = (bf16*)alloc((size_t)1024 * 1024 * 2);
  bf16* Wh1gT = (bf16*)alloc((size_t)1024 * 1024 * 2);
  bf16* WyT   = (bf16*)alloc((size_t)256 * 1024 * 2);
  bf16* xb    = (bf16*)alloc((size_t)BB * SS * II * 2);
  float* h0f  = (float*)alloc((size_t)BB * HH * 4);
  float* h1f  = (float*)alloc((size_t)BB * HH * 4);
  bf16* h0b   = (bf16*)alloc((size_t)BB * HH * 2);
  bf16* h1b   = (bf16*)alloc((size_t)BB * HH * 2);
  float* z0b  = (float*)alloc((size_t)BB * HH * 4);
  float* xg0  = (float*)alloc((size_t)BB * HH * 4);
  bf16* rh0b  = (bf16*)alloc((size_t)BB * HH * 2);
  float* z1b  = (float*)alloc((size_t)BB * HH * 4);
  float* xg1  = (float*)alloc((size_t)BB * HH * 4);
  bf16* rh1b  = (bf16*)alloc((size_t)BB * HH * 2);

  hipMemsetAsync(bar, 0, 256, stream);

  dim3 tb(32, 8);
  // WzrT0[n][k]: k<256 -> Wx0[k][n], else Wh0[k-256][n]   (n in [0,2048): z|r columns)
  pack_wT<<<dim3(256 / 32, 2048 / 32), tb, 0, stream>>>(Wx0, 3 * HH, 0, WzrT0, II + HH, 0);
  pack_wT<<<dim3(1024 / 32, 2048 / 32), tb, 0, stream>>>(Wh0, 3 * HH, 0, WzrT0, II + HH, II);
  pack_wT<<<dim3(256 / 32, 1024 / 32), tb, 0, stream>>>(Wx0, 3 * HH, 2 * HH, Wg0xT, II, 0);
  pack_wT<<<dim3(1024 / 32, 1024 / 32), tb, 0, stream>>>(Wh0, 3 * HH, 2 * HH, Wh0gT, HH, 0);
  pack_wT<<<dim3(1024 / 32, 2048 / 32), tb, 0, stream>>>(Wx1, 3 * HH, 0, W1zrT, 2 * HH, 0);
  pack_wT<<<dim3(1024 / 32, 2048 / 32), tb, 0, stream>>>(Wh1, 3 * HH, 0, W1zrT, 2 * HH, HH);
  pack_wT<<<dim3(1024 / 32, 1024 / 32), tb, 0, stream>>>(Wx1, 3 * HH, 2 * HH, Wx1gT, HH, 0);
  pack_wT<<<dim3(1024 / 32, 1024 / 32), tb, 0, stream>>>(Wh1, 3 * HH, 2 * HH, Wh1gT, HH, 0);
  pack_wT<<<dim3(1024 / 32, 256 / 32), tb, 0, stream>>>(Wy, OO, 0, WyT, HH, 0);

  const int nx = BB * SS * II;
  pack_bf16<<<(nx + 255) / 256, 256, 0, stream>>>(x, xb, nx);
  init_h<<<(BB * HH + 255) / 256, 256, 0, stream>>>(h0, h0f, h1f, h0b, h1b);

  gru_persistent<<<NWG, 256, 0, stream>>>(xb, WzrT0, Wg0xT, Wh0gT, W1zrT, Wx1gT, Wh1gT, WyT,
                                          bh0, bh1, by, h0f, h1f, h0b, h1b,
                                          z0b, xg0, rh0b, z1b, xg1, rh1b, out, bar);
}
